// RPNModule_67448166416469
// MI455X (gfx1250) — matrix operand order repaired and checked by reading, not compiled, not run on hardware
//
#include <hip/hip_runtime.h>
#include <hip/hip_bf16.h>
#include <stdint.h>

typedef __bf16 bf16_t;
typedef __attribute__((ext_vector_type(16))) __bf16 v16bf;
typedef __attribute__((ext_vector_type(8)))  __bf16 v8bf;
typedef __attribute__((ext_vector_type(8)))  float  v8f;
typedef __attribute__((ext_vector_type(4)))  unsigned int u32x4;
typedef __attribute__((ext_vector_type(8)))  int i32x8;
typedef __attribute__((ext_vector_type(4)))  int i32x4;

#define SUM_A   261888      // 196608+49152+12288+3072+768
#define K_ALL   4768        // 1000*4 + 768
#define BBOX_CLIP 4.135166556742356f
#define NEG_INF (-3.402823e38f)

__device__ const int   c_A[5]    = {196608, 49152, 12288, 3072, 768};
__device__ const int   c_OFFA[5] = {0, 196608, 245760, 258048, 261120};
__device__ const int   c_K[5]    = {1000, 1000, 1000, 1000, 768};
__device__ const int   c_LOFF[5] = {0, 1000, 2000, 3000, 4000};
__device__ const int   c_H[5]    = {256, 128, 64, 32, 16};
__device__ const int   c_SIZE[5] = {32, 64, 128, 256, 512};
__device__ const float c_RATIO[3] = {0.5f, 1.0f, 2.0f};

__device__ __forceinline__ unsigned fkey(float f) {
    unsigned u = __float_as_uint(f);
    return (u & 0x80000000u) ? ~u : (u | 0x80000000u);
}

__device__ __forceinline__ v16bf cat16(v8bf lo, v8bf hi) {
    return __builtin_shufflevector(lo, hi, 0,1,2,3,4,5,6,7,8,9,10,11,12,13,14,15);
}

// ---------------------------------------------------------------------------
// 0) Weight repack: conv_w[ko][c][3][3] fp32 -> Wr[tap][ko][c] bf16 so conv A
//    staging is 16B-contiguous; cls/bbox 1x1 -> fused 16x256 bf16 head A.
// ---------------------------------------------------------------------------
__global__ __launch_bounds__(256) void repack_weights(
    const float* __restrict__ conv_w, const float* __restrict__ cls_w,
    const float* __restrict__ bbox_w, bf16_t* __restrict__ Wr,
    bf16_t* __restrict__ Hw) {
    int i = blockIdx.x * 256 + threadIdx.x;
    if (i < 9 * 256 * 256) {
        int c = i & 255, ko = (i >> 8) & 255, tap = i >> 16;
        Wr[i] = (bf16_t)conv_w[((size_t)ko * 256 + c) * 9 + tap];
    }
    if (i < 16 * 256) {
        int c = i & 255, m = i >> 8;
        float v = 0.0f;
        if (m < 3)       v = cls_w[m * 256 + c];
        else if (m < 15) v = bbox_w[(m - 3) * 256 + c];
        Hw[i] = (bf16_t)v;
    }
}

// ---------------------------------------------------------------------------
// 1) 3x3 conv 256->256 + bias + ReLU: implicit GEMM on v_wmma_f32_16x16x32_bf16.
//    WG = 256 thr (8 wave32), tile = 64 out-ch x 64 px, K = 2304 in 72 steps,
//    double-buffered LDS (1 barrier/step), 2 WMMAs per wave per step.
// ---------------------------------------------------------------------------
__global__ __launch_bounds__(256) void conv3x3_wmma(
    const float* __restrict__ feat, const bf16_t* __restrict__ Wr,
    const float* __restrict__ conv_b, bf16_t* __restrict__ tlvl,
    size_t imgStride, int H) {
    __shared__ __align__(16) bf16_t Ab[2][64 * 40];   // [m][k], pitch 80B
    __shared__ __align__(16) bf16_t Bb[2][64 * 40];   // [px][k], pitch 80B

    const int tid = threadIdx.x, lane = tid & 31, wv = tid >> 5;
    const int mi = wv >> 1, nq = (wv & 1) * 2;        // wave: m-subtile mi, n-subtiles nq,nq+1
    const int p0 = blockIdx.x * 64, kt = blockIdx.y, n = blockIdx.z;
    const int HW = H * H;

    // A staging: one 8-half chunk per thread (global b128 -> LDS b128)
    const int am = tid >> 2, aj = (tid & 3) * 8;
    // B staging: 8 scalar gathers per thread (transpose into [px][k])
    const int bpx = tid & 63, bj0 = tid >> 6;
    const int bp = p0 + bpx, bx = bp % H, by = bp / H;

    v8f acc0 = {}, acc1 = {};

    auto stageA = [&](int buf, int tap, int cb) {
        v8bf w8 = *(const v8bf*)&Wr[((size_t)tap * 256 + kt * 64 + am) * 256 + cb + aj];
        *(v8bf*)&Ab[buf][am * 40 + aj] = w8;
    };
    auto stageB = [&](int buf, int tap, int cb) {
        const int dy = tap / 3 - 1, dx = tap % 3 - 1;
        const int xg = bx + dx, yg = by + dy;
        const bool inb = (xg >= 0) & (xg < H) & (yg >= 0) & (yg < H);
#pragma unroll
        for (int r = 0; r < 8; ++r) {
            int j = bj0 + 4 * r;
            float v = inb ? feat[(((size_t)n * 256 + cb + j) * H + yg) * H + xg] : 0.0f;
            Bb[buf][bpx * 40 + j] = (bf16_t)v;
        }
    };

    stageA(0, 0, 0);
    stageB(0, 0, 0);

    for (int ks = 0; ks < 72; ++ks) {
        const int buf = ks & 1;
        __syncthreads();
        if (ks + 1 < 72) {
            const int t2 = (ks + 1) >> 3, c2 = ((ks + 1) & 7) * 32;
            stageA(buf ^ 1, t2, c2);
            stageB(buf ^ 1, t2, c2);
            if (ks + 2 < 72) {    // L2 warm-up two steps ahead
                const int t3 = (ks + 2) >> 3, c3 = ((ks + 2) & 7) * 32;
                const int xg = bx + (t3 % 3 - 1), yg = by + (t3 / 3 - 1);
                if ((xg >= 0) & (xg < H) & (yg >= 0) & (yg < H))
                    __builtin_prefetch(&feat[(((size_t)n * 256 + c3 + bj0) * H + yg) * H + xg], 0, 1);
            }
        }
        // A fragment (16x32 bf16 per ISA lane layout; two 16B LDS loads)
        const int mrow = mi * 16 + (lane & 15);
        const int kh = (lane >> 4) * 8;
        v8bf alo = *(const v8bf*)&Ab[buf][mrow * 40 + kh];
        v8bf ahi = *(const v8bf*)&Ab[buf][mrow * 40 + 16 + kh];
        v16bf af = cat16(alo, ahi);
        // B fragments for the wave's two n-subtiles
        const int kg = (lane >> 4) * 8;
        const int nc0 = (nq + 0) * 16 + (lane & 15);
        const int nc1 = (nq + 1) * 16 + (lane & 15);
        v16bf bf0 = cat16(*(const v8bf*)&Bb[buf][nc0 * 40 + kg],
                          *(const v8bf*)&Bb[buf][nc0 * 40 + kg + 16]);
        v16bf bf1 = cat16(*(const v8bf*)&Bb[buf][nc1 * 40 + kg],
                          *(const v8bf*)&Bb[buf][nc1 * 40 + kg + 16]);
        acc0 = __builtin_amdgcn_wmma_f32_16x16x32_bf16(
            false, af, false, bf0, (short)0, acc0, false, false);
        acc1 = __builtin_amdgcn_wmma_f32_16x16x32_bf16(
            false, af, false, bf1, (short)0, acc1, false, false);
    }
    // epilogue: bias + ReLU, store bf16 t
    const size_t base = (size_t)n * imgStride;
    const int pxo0 = p0 + (nq + 0) * 16 + (lane & 15);
    const int pxo1 = p0 + (nq + 1) * 16 + (lane & 15);
#pragma unroll
    for (int r = 0; r < 8; ++r) {
        int c = kt * 64 + mi * 16 + r + (lane >> 4) * 8;
        float bsv = conv_b[c];
        float v0 = fmaxf(acc0[r] + bsv, 0.0f);
        float v1 = fmaxf(acc1[r] + bsv, 0.0f);
        tlvl[base + (size_t)c * HW + pxo0] = (bf16_t)v0;
        tlvl[base + (size_t)c * HW + pxo1] = (bf16_t)v1;
    }
}

// ---------------------------------------------------------------------------
// 2) 1x1 heads: fused 16(out)x128(px) WMMA GEMM over K=256 from bf16 t.
//    The 16x256 weight tile is ALSO issued through the Tensor Data Mover
//    (tensor_load_to_lds + s_wait_tensorcnt) to exercise the CDNA5 TDM path.
// ---------------------------------------------------------------------------
__global__ __launch_bounds__(256) void heads_wmma(
    const bf16_t* __restrict__ tlvl, size_t imgStride,
    const bf16_t* __restrict__ Hw, const float* __restrict__ cls_b,
    const float* __restrict__ bbox_b, float* __restrict__ S,
    float* __restrict__ D4, int H, int offA) {
    __shared__ __align__(16) bf16_t Ah[16 * 264];  // [m][c], pitch 528B (256 + 4 DW pad)
    __shared__ __align__(16) bf16_t Bh[128 * 40];  // [px][k], pitch 80B

    const int tid = threadIdx.x, lane = tid & 31, wv = tid >> 5;
    const int n = blockIdx.y, p0 = blockIdx.x * 128, HW = H * H;
    const bf16_t* tb = tlvl + (size_t)n * imgStride;

    // --- TDM: 2D tile 16 rows x 256 cols of 2B data, LDS rows padded +4 DW.
    if (wv == 0) {
        unsigned long long ga = (unsigned long long)(uintptr_t)Hw;
        unsigned lds = (unsigned)(uintptr_t)&Ah[0];
        u32x4 g0 = {1u,                                   // count=1 (valid D#)
                    lds,                                  // lds_addr
                    (unsigned)(ga & 0xFFFFFFFFu),         // global_addr lo
                    (unsigned)((ga >> 32) & 0x01FFFFFFu) | 0x80000000u}; // hi | type=2
        i32x8 g1 = {0x07910000,   // data_size=2B, pad_enable, interval=128DW, amount=4DW
                    0x01000000,   // tensor_dim0 = 256 (bits 63:48)
                    0x00100000,   // tensor_dim1 = 16  (bits 95:80)
                    0x01000000,   // tile_dim0   = 256 (bits 127:112)
                    16,           // tile_dim1   = 16
                    256,          // tensor_dim0_stride = 256
                    0, 0};
        i32x4 gz4 = {0, 0, 0, 0};
        i32x8 gz8 = {0, 0, 0, 0, 0, 0, 0, 0};
        __builtin_amdgcn_tensor_load_to_lds(g0, g1, gz4, gz4, gz8, 0);
    }
    __builtin_amdgcn_s_wait_tensorcnt(0);
    // Redundant plain staging of the same tile (keeps semantics independent
    // of the experimental TDM descriptor).
#pragma unroll
    for (int r = 0; r < 16; ++r) {
        int lin = tid + 256 * r;
        int c = lin & 255, m = lin >> 8;
        Ah[m * 264 + c] = Hw[m * 256 + c];
    }
    v8f acc = {};
    for (int ks = 0; ks < 8; ++ks) {
        __syncthreads();
        const int px = tid & 127, c0 = tid >> 7;
#pragma unroll
        for (int r = 0; r < 16; ++r) {
            int c = c0 + 2 * r;
            Bh[px * 40 + c] = tb[(size_t)(ks * 32 + c) * HW + p0 + px];
        }
        __syncthreads();
        const int m = lane & 15, kh = (lane >> 4) * 8;
        v8bf alo = *(const v8bf*)&Ah[m * 264 + ks * 32 + kh];
        v8bf ahi = *(const v8bf*)&Ah[m * 264 + ks * 32 + 16 + kh];
        v16bf af = cat16(alo, ahi);
        const int ncol = wv * 16 + (lane & 15);
        const int kg = (lane >> 4) * 8;
        v8bf blo = *(const v8bf*)&Bh[ncol * 40 + kg];
        v8bf bhi = *(const v8bf*)&Bh[ncol * 40 + kg + 16];
        v16bf bfr = cat16(blo, bhi);
        acc = __builtin_amdgcn_wmma_f32_16x16x32_bf16(
            false, af, false, bfr, (short)0, acc, false, false);
    }
    const int px = p0 + wv * 16 + (lane & 15);
#pragma unroll
    for (int r = 0; r < 8; ++r) {
        int m = r + (lane >> 4) * 8;
        float v = acc[r];
        if (m < 3) {
            S[(size_t)n * SUM_A + offA + (size_t)px * 3 + m] = v + cls_b[m];
        } else if (m < 15) {
            int q = m - 3, a = q >> 2, c = q & 3;
            D4[((size_t)n * SUM_A + offA + (size_t)px * 3 + a) * 4 + c] = v + bbox_b[q];
        }
    }
}

// ---------------------------------------------------------------------------
// 3) Per-(image,level) top-k candidate selection: 256-bin radix histogram on
//    the top byte of the order-preserving key, compact <=2048 candidates.
// ---------------------------------------------------------------------------
__global__ __launch_bounds__(256) void topk_hist(
    const float* __restrict__ S, float* __restrict__ candK,
    int* __restrict__ candI) {
    __shared__ unsigned hist[256];
    __shared__ int sThr, sHi, cntHi, cntEq;
    const int l = blockIdx.x, n = blockIdx.y, tid = threadIdx.x;
    const int A = c_A[l], k = c_K[l];
    const size_t sbase = (size_t)n * SUM_A + c_OFFA[l];
    const int cbase = (n * 5 + l) * 2048;

    hist[tid] = 0u;
    for (int i = tid; i < 2048; i += 256) { candK[cbase + i] = NEG_INF; candI[cbase + i] = 0; }
    __syncthreads();
    for (int i = tid; i < A; i += 256)
        atomicAdd(&hist[fkey(S[sbase + i]) >> 24], 1u);
    __syncthreads();
    if (tid == 0) {
        int cum = 0, thr = 0, hi = 0;
        for (int b = 255; b >= 0; --b) {
            int c = (int)hist[b];
            if (cum + c >= k) { thr = b; hi = cum; break; }
            cum += c;
        }
        sThr = thr; sHi = hi; cntHi = 0; cntEq = 0;
    }
    __syncthreads();
    const int thr = sThr, hi = sHi;
    for (int i = tid; i < A; i += 256) {
        float s = S[sbase + i];
        int b = (int)(fkey(s) >> 24);
        if (b > thr) {
            int slot = atomicAdd(&cntHi, 1);
            candK[cbase + slot] = s; candI[cbase + slot] = i;
        } else if (b == thr) {
            int slot = hi + atomicAdd(&cntEq, 1);
            if (slot < 2048) { candK[cbase + slot] = s; candI[cbase + slot] = i; }
        }
    }
}

// ---------------------------------------------------------------------------
// 4) Sort 2048 candidates (bitonic, desc), then anchors + decode + clip +
//    min-size mask + sigmoid into the per-image concatenated K=4768 list.
// ---------------------------------------------------------------------------
__global__ __launch_bounds__(1024) void sortsel(
    const float* __restrict__ candK, const int* __restrict__ candI,
    const float* __restrict__ D4, float* __restrict__ CB,
    float* __restrict__ CS) {
    __shared__ float ky[2048];
    __shared__ int   id[2048];
    const int l = blockIdx.x, n = blockIdx.y, tid = threadIdx.x;
    const int cbase = (n * 5 + l) * 2048;
    for (int i = tid; i < 2048; i += 1024) { ky[i] = candK[cbase + i]; id[i] = candI[cbase + i]; }
    __syncthreads();
    for (int sz = 2; sz <= 2048; sz <<= 1)
        for (int st = sz >> 1; st > 0; st >>= 1) {
            for (int i = tid; i < 2048; i += 1024) {
                int j = i ^ st;
                if (j > i) {
                    bool dsc = ((i & sz) == 0);
                    float a = ky[i], b = ky[j];
                    if (dsc ? (a < b) : (a > b)) {
                        ky[i] = b; ky[j] = a;
                        int t = id[i]; id[i] = id[j]; id[j] = t;
                    }
                }
            }
            __syncthreads();
        }
    const int k = c_K[l], W = c_H[l], offA = c_OFFA[l];
    const float stride = 1024.0f / (float)W;
    for (int i = tid; i < k; i += 1024) {
        int slot = n * K_ALL + c_LOFF[l] + i;
        float key = ky[i];
        float x0 = 0.f, y0 = 0.f, x1 = 0.f, y1 = 0.f, sc = -1.0f;
        if (key > -3.0e38f) {
            int idx = id[i];
            int a = idx % 3, cell = idx / 3;
            int gx = cell % W, gy = cell / W;
            float hr = sqrtf(c_RATIO[a]), wr = 1.0f / hr;
            float sz = (float)c_SIZE[l];
            float ws = wr * sz, hs = hr * sz;
            float sx = gx * stride, sy = gy * stride;
            float a0 = roundf(-0.5f * ws) + sx, a1 = roundf(-0.5f * hs) + sy;
            float a2 = roundf( 0.5f * ws) + sx, a3 = roundf( 0.5f * hs) + sy;
            float aw = a2 - a0, ah = a3 - a1;
            float acx = a0 + 0.5f * aw, acy = a1 + 0.5f * ah;
            size_t db = ((size_t)n * SUM_A + offA + idx) * 4;
            float dx = D4[db], dyv = D4[db + 1];
            float dw = fminf(D4[db + 2], BBOX_CLIP), dh = fminf(D4[db + 3], BBOX_CLIP);
            float pcx = dx * aw + acx, pcy = dyv * ah + acy;
            float pw = expf(dw) * aw, ph = expf(dh) * ah;
            x0 = fminf(fmaxf(pcx - 0.5f * pw, 0.f), 1024.f);
            y0 = fminf(fmaxf(pcy - 0.5f * ph, 0.f), 1024.f);
            x1 = fminf(fmaxf(pcx + 0.5f * pw, 0.f), 1024.f);
            y1 = fminf(fmaxf(pcy + 0.5f * ph, 0.f), 1024.f);
            bool valid = ((x1 - x0) >= 1e-3f) && ((y1 - y0) >= 1e-3f);
            sc = valid ? 1.0f / (1.0f + expf(-key)) : -1.0f;
        }
        CB[(size_t)slot * 4 + 0] = x0; CB[(size_t)slot * 4 + 1] = y0;
        CB[(size_t)slot * 4 + 2] = x1; CB[(size_t)slot * 4 + 3] = y1;
        CS[slot] = sc;
    }
}

// ---------------------------------------------------------------------------
// 5) Per-image global sort of the 4768 masked scores (bitonic over 8192).
// ---------------------------------------------------------------------------
__global__ __launch_bounds__(1024) void gsort(
    const float* __restrict__ CS, float* __restrict__ SS, int* __restrict__ ORD) {
    __shared__ float ky[8192];
    __shared__ int   id[8192];
    const int n = blockIdx.x, tid = threadIdx.x;
    for (int i = tid; i < 8192; i += 1024) {
        ky[i] = (i < K_ALL) ? CS[n * K_ALL + i] : NEG_INF;
        id[i] = (i < K_ALL) ? i : 0;
    }
    __syncthreads();
    for (int sz = 2; sz <= 8192; sz <<= 1)
        for (int st = sz >> 1; st > 0; st >>= 1) {
            for (int i = tid; i < 8192; i += 1024) {
                int j = i ^ st;
                if (j > i) {
                    bool dsc = ((i & sz) == 0);
                    float a = ky[i], b = ky[j];
                    if (dsc ? (a < b) : (a > b)) {
                        ky[i] = b; ky[j] = a;
                        int t = id[i]; id[i] = id[j]; id[j] = t;
                    }
                }
            }
            __syncthreads();
        }
    for (int i = tid; i < K_ALL; i += 1024) {
        SS[n * K_ALL + i] = ky[i];
        ORD[n * K_ALL + i] = id[i];
    }
}

// ---------------------------------------------------------------------------
// 6) Greedy NMS (per-level 1025 offsets) + top-1000 compaction to d_out.
// ---------------------------------------------------------------------------
__global__ __launch_bounds__(1024) void nms_kernel(
    const float* __restrict__ CB, const float* __restrict__ SS,
    const int* __restrict__ ORD, float* __restrict__ BS,
    float* __restrict__ out) {
    __shared__ int alive[K_ALL];
    const int n = blockIdx.x, tid = threadIdx.x;
    for (int r = tid; r < K_ALL; r += 1024) {
        int j = ORD[n * K_ALL + r];
        int lvl = j / 1000; if (lvl > 4) lvl = 4;
        float offv = lvl * 1025.0f;
        const float* cb = &CB[((size_t)n * K_ALL + j) * 4];
        float* bs = &BS[((size_t)n * K_ALL + r) * 4];
        bs[0] = cb[0] + offv; bs[1] = cb[1] + offv;
        bs[2] = cb[2] + offv; bs[3] = cb[3] + offv;
        alive[r] = SS[n * K_ALL + r] > -0.5f;
    }
    __syncthreads();
    for (int i = 0; i < K_ALL; ++i) {
        bool ai = alive[i] != 0;
        if (ai) {
            const float* bi = &BS[((size_t)n * K_ALL + i) * 4];
            float bx0 = bi[0], by0 = bi[1], bx1 = bi[2], by1 = bi[3];
            float areai = (bx1 - bx0) * (by1 - by0);
            for (int j = i + 1 + tid; j < K_ALL; j += 1024) {
                if (alive[j]) {
                    const float* bj = &BS[((size_t)n * K_ALL + j) * 4];
                    float xx0 = fmaxf(bx0, bj[0]), yy0 = fmaxf(by0, bj[1]);
                    float xx1 = fminf(bx1, bj[2]), yy1 = fminf(by1, bj[3]);
                    float w = fmaxf(xx1 - xx0, 0.f), h = fmaxf(yy1 - yy0, 0.f);
                    float inter = w * h;
                    float areaj = (bj[2] - bj[0]) * (bj[3] - bj[1]);
                    if (inter / (areai + areaj - inter) > 0.7f) alive[j] = 0;
                }
            }
        }
        __syncthreads();
    }
    if (tid == 0) {
        int cnt = 0;
        for (int i = 0; i < K_ALL && cnt < 1000; ++i) {
            if (alive[i]) {
                int j = ORD[n * K_ALL + i];
                const float* cb = &CB[((size_t)n * K_ALL + j) * 4];
                float* ob = &out[((size_t)n * 1000 + cnt) * 4];
                ob[0] = cb[0]; ob[1] = cb[1]; ob[2] = cb[2]; ob[3] = cb[3];
                out[8000 + n * 1000 + cnt] = SS[n * K_ALL + i];
                ++cnt;
            }
        }
    }
}

// ---------------------------------------------------------------------------
extern "C" void kernel_launch(void* const* d_in, const int* in_sizes, int n_in,
                              void* d_out, int out_size, void* d_ws, size_t ws_size,
                              hipStream_t stream) {
    const float* feats[5] = {(const float*)d_in[1], (const float*)d_in[2],
                             (const float*)d_in[3], (const float*)d_in[4],
                             (const float*)d_in[5]};
    const float* conv_w = (const float*)d_in[6];
    const float* conv_b = (const float*)d_in[7];
    const float* cls_w  = (const float*)d_in[8];
    const float* cls_b  = (const float*)d_in[9];
    const float* bbox_w = (const float*)d_in[10];
    const float* bbox_b = (const float*)d_in[11];

    static const int Hs[5] = {256, 128, 64, 32, 16};
    static const size_t tl[5] = {0, 16777216, 20971520, 22020096, 22282240};
    static const int offA[5] = {0, 196608, 245760, 258048, 261120};
    const size_t TSZ = 22347776;  // per-image bf16 t elements

    char* w = (char*)d_ws;
    auto carve = [&](size_t bytes) -> char* {
        char* p = w; w += (bytes + 255) & ~size_t(255); return p;
    };
    bf16_t* T     = (bf16_t*)carve(2 * TSZ * sizeof(bf16_t));
    bf16_t* Wr    = (bf16_t*)carve(9 * 256 * 256 * sizeof(bf16_t));
    bf16_t* Hw    = (bf16_t*)carve(16 * 256 * sizeof(bf16_t));
    float*  Sbuf  = (float*)carve((size_t)2 * SUM_A * sizeof(float));
    float*  D4    = (float*)carve((size_t)2 * SUM_A * 4 * sizeof(float));
    float*  candK = (float*)carve((size_t)2 * 5 * 2048 * sizeof(float));
    int*    candI = (int*)carve((size_t)2 * 5 * 2048 * sizeof(int));
    float*  CB    = (float*)carve((size_t)2 * K_ALL * 4 * sizeof(float));
    float*  CS    = (float*)carve((size_t)2 * K_ALL * sizeof(float));
    float*  SS    = (float*)carve((size_t)2 * K_ALL * sizeof(float));
    int*    ORD   = (int*)carve((size_t)2 * K_ALL * sizeof(int));
    float*  BS    = (float*)carve((size_t)2 * K_ALL * 4 * sizeof(float));

    (void)hipMemsetAsync(d_out, 0, (size_t)out_size * sizeof(float), stream);

    repack_weights<<<(9 * 256 * 256 + 255) / 256, 256, 0, stream>>>(
        conv_w, cls_w, bbox_w, Wr, Hw);

    for (int l = 0; l < 5; ++l) {
        int H = Hs[l], HW = H * H;
        conv3x3_wmma<<<dim3(HW / 64, 4, 2), 256, 0, stream>>>(
            feats[l], Wr, conv_b, T + tl[l], TSZ, H);
    }
    for (int l = 0; l < 5; ++l) {
        int H = Hs[l], HW = H * H;
        heads_wmma<<<dim3(HW / 128, 2), 256, 0, stream>>>(
            T + tl[l], TSZ, Hw, cls_b, bbox_b, Sbuf, D4, H, offA[l]);
    }
    topk_hist<<<dim3(5, 2), 256, 0, stream>>>(Sbuf, candK, candI);
    sortsel<<<dim3(5, 2), 1024, 0, stream>>>(candK, candI, D4, CB, CS);
    gsort<<<2, 1024, 0, stream>>>(CS, SS, ORD);
    nms_kernel<<<2, 1024, 0, stream>>>(CB, SS, ORD, BS, (float*)d_out);
}
